// GraphSAGE_90142773609390
// MI455X (gfx1250) — matrix operand order repaired and checked by reading, not compiled, not run on hardware
//
#include <hip/hip_runtime.h>
#include <hip/hip_bf16.h>

// ---------------------------------------------------------------------------
// GraphSAGE (B=1024, S=32, N=100000, IN=128, H=256, OUT=128), eval mode.
// Decomposition:  max_j([src;n_j]@W + b) = src@W_top + b + max_j(n_j@W_bot)
// => precompute y = x @ [W1_top | W1_bot]  ([N,512], bf16, L2-resident),
//    then gather+max, then small layer-2 GEMMs, combine, output projection.
// GEMMs use v_wmma_f32_16x16x32_bf16 (wave32, f32 accum), register
// double-buffered K-pipeline, 32x32 output block per wave (spill-free).
// ---------------------------------------------------------------------------

#define Bsz   1024
#define Ssz   32
#define Nnod  100000
#define INF_  128
#define Hdim  256
#define OUTD  128

typedef __attribute__((ext_vector_type(16))) __bf16 v16bf;
typedef __attribute__((ext_vector_type(8)))  float  v8f;

union Frag16 { v16bf v; uint4 q[2]; };

__device__ __forceinline__ float bf_lo(unsigned int u) {
    return __uint_as_float(u << 16);
}
__device__ __forceinline__ float bf_hi(unsigned int u) {
    return __uint_as_float(u & 0xffff0000u);
}

// ---- elementwise f32 -> bf16 ------------------------------------------------
__global__ void cvt_bf16_kernel(const float* __restrict__ src,
                                __hip_bfloat16* __restrict__ dst, int n) {
    int i = blockIdx.x * blockDim.x + threadIdx.x;
    if (i < n) dst[i] = __float2bfloat16(src[i]);
}

// ---- transpose-convert: src f32 [rows, cols] -> dst bf16 [cols, rows] ------
__global__ void transpose_bf16_kernel(const float* __restrict__ src,
                                      __hip_bfloat16* __restrict__ dst,
                                      int rows, int cols) {
    int i = blockIdx.x * blockDim.x + threadIdx.x;
    if (i >= rows * cols) return;
    int r = i / cols, c = i % cols;
    dst[(size_t)c * rows + r] = __float2bfloat16(src[(size_t)r * cols + c]);
}

// ---- bf16 GEMM via WMMA: C[M,Nc] = A[M,K] @ Bt[Nc,K]^T ---------------------
// One wave per block; wave computes a 32(M) x 32(N) block (2x2 WMMA tiles).
// grid.x = M/32, grid.y = Nc/32.  Requires K % 64 == 0.
// Two-stage register double buffer: stage s loads overlap stage s^1 WMMAs.
// Register budget: 4 acc (32) + 8 frags (64) + addressing  -> no spills.
__global__ void __launch_bounds__(32)
gemm_bf16_wmma(const __hip_bfloat16* __restrict__ A,
               const __hip_bfloat16* __restrict__ Bt,
               float* __restrict__ Cf,            // f32 out (or null)
               __hip_bfloat16* __restrict__ Cb,   // bf16 out (or null)
               const float* __restrict__ bias,    // [Nc] or null
               int K, int ldc) {
    const int lane = threadIdx.x;      // 0..31
    const int m0   = blockIdx.x * 32;
    const int n0   = blockIdx.y * 32;
    const int mrow = lane & 15;        // A row within tile / B column / C column
    const int half = lane >> 4;

    const __hip_bfloat16* aRow0 = A + (size_t)(m0 + mrow) * K + 8 * half;
    const __hip_bfloat16* aRow1 = aRow0 + (size_t)16 * K;
    const __hip_bfloat16* bCol0 = Bt + (size_t)(n0 + mrow) * K + 8 * half;
    const __hip_bfloat16* bCol1 = bCol0 + (size_t)16 * K;

    v8f acc[2][2] = {{v8f{}, v8f{}}, {v8f{}, v8f{}}};

    Frag16 aF[2][2];   // [stage][m-tile]
    Frag16 bF[2][2];   // [stage][n-tile]

    // A layout (16-bit, 16x32): lane-half 0 holds K {0..7, 16..23},
    // half 1 holds K {8..15, 24..31} -> two contiguous 16B chunks at +0, +16.
    // B layout (16-bit, 32x16): lane = column, same K split as A
    // -> two contiguous 16B chunks at +0, +16.
#define LOAD_STAGE(s, kk)                                                     \
    do {                                                                      \
        aF[s][0].q[0] = *(const uint4*)(aRow0 + (kk));                        \
        aF[s][0].q[1] = *(const uint4*)(aRow0 + (kk) + 16);                   \
        aF[s][1].q[0] = *(const uint4*)(aRow1 + (kk));                        \
        aF[s][1].q[1] = *(const uint4*)(aRow1 + (kk) + 16);                   \
        bF[s][0].q[0] = *(const uint4*)(bCol0 + (kk));                        \
        bF[s][0].q[1] = *(const uint4*)(bCol0 + (kk) + 16);                   \
        bF[s][1].q[0] = *(const uint4*)(bCol1 + (kk));                        \
        bF[s][1].q[1] = *(const uint4*)(bCol1 + (kk) + 16);                   \
    } while (0)

#define MMA_STAGE(s)                                                          \
    do {                                                                      \
        acc[0][0] = __builtin_amdgcn_wmma_f32_16x16x32_bf16(                  \
            false, aF[s][0].v, false, bF[s][0].v, (short)0, acc[0][0],        \
            false, false);                                                    \
        acc[0][1] = __builtin_amdgcn_wmma_f32_16x16x32_bf16(                  \
            false, aF[s][0].v, false, bF[s][1].v, (short)0, acc[0][1],        \
            false, false);                                                    \
        acc[1][0] = __builtin_amdgcn_wmma_f32_16x16x32_bf16(                  \
            false, aF[s][1].v, false, bF[s][0].v, (short)0, acc[1][0],        \
            false, false);                                                    \
        acc[1][1] = __builtin_amdgcn_wmma_f32_16x16x32_bf16(                  \
            false, aF[s][1].v, false, bF[s][1].v, (short)0, acc[1][1],        \
            false, false);                                                    \
    } while (0)

    LOAD_STAGE(0, 0);
    for (int k0 = 0; k0 < K; k0 += 64) {
        LOAD_STAGE(1, k0 + 32);     // in flight during stage-0 WMMAs
        MMA_STAGE(0);
        if (k0 + 64 < K) LOAD_STAGE(0, k0 + 64);  // in flight during stage-1
        MMA_STAGE(1);
    }
#undef LOAD_STAGE
#undef MMA_STAGE

    // C/D layout (32-bit, 16x16): VGPR r holds row r (lanes 0-15) / r+8
    // (lanes 16-31), column = lane & 15.
    const int rowOff = half * 8;
#pragma unroll
    for (int mi = 0; mi < 2; ++mi) {
#pragma unroll
        for (int t = 0; t < 2; ++t) {
            const int col = n0 + 16 * t + mrow;
            const float bv = bias ? bias[col] : 0.0f;
#pragma unroll
            for (int r = 0; r < 8; ++r) {
                float v = acc[mi][t][r] + bv;
                size_t idx = (size_t)(m0 + 16 * mi + rowOff + r) * ldc + col;
                if (Cf) Cf[idx] = v;
                else    Cb[idx] = __float2bfloat16(v);
            }
        }
    }
}

// ---- layer-1 gather + max + bias + relu ------------------------------------
// out[g, :] = relu( y_top[srcIdx[g]] + b1 + max_j y_bot[nbrIdx[g*32 + j]] )
// y is [Nnod, 512] bf16: cols 0..255 = y_top, 256..511 = y_bot.
// block = 128 threads, each handles 2 adjacent columns (one dword).
__global__ void sage1_pool_kernel(const __hip_bfloat16* __restrict__ y,
                                  const int* __restrict__ srcIdx,
                                  const int* __restrict__ nbrIdx,
                                  const float* __restrict__ b1,
                                  __hip_bfloat16* __restrict__ out) {
    const int g = blockIdx.x;
    const int t = threadIdx.x;  // 0..127
    const int* idx = nbrIdx + (size_t)g * Ssz;

    // Block-uniform indices -> scalar loads; preload so the 32 row-gathers
    // can be issued in independent batches.
    int nodes[Ssz];
#pragma unroll
    for (int j = 0; j < Ssz; ++j) nodes[j] = idx[j];

    float m0 = -INFINITY, m1 = -INFINITY;
#pragma unroll 8
    for (int j = 0; j < Ssz; ++j) {
        unsigned int u =
            *(const unsigned int*)(y + (size_t)nodes[j] * 512 + 256 + 2 * t);
        m0 = fmaxf(m0, bf_lo(u));
        m1 = fmaxf(m1, bf_hi(u));
    }
    const int sn = srcIdx[g];
    unsigned int ut = *(const unsigned int*)(y + (size_t)sn * 512 + 2 * t);
    float v0 = fmaxf(bf_lo(ut) + b1[2 * t]     + m0, 0.0f);
    float v1 = fmaxf(bf_hi(ut) + b1[2 * t + 1] + m1, 0.0f);
    out[(size_t)g * Hdim + 2 * t]     = __float2bfloat16(v0);
    out[(size_t)g * Hdim + 2 * t + 1] = __float2bfloat16(v1);
}

// ---- layer-2 combine: h2[b,h] = zsrc[b,h] + b2[h] + max_s z[b*S+s, h] ------
__global__ void sage2_combine_kernel(const float* __restrict__ z,
                                     const float* __restrict__ zsrc,
                                     const float* __restrict__ b2,
                                     __hip_bfloat16* __restrict__ h2b) {
    const int b = blockIdx.x;
    const int t = threadIdx.x;  // 0..255
    float m = -INFINITY;
#pragma unroll 8
    for (int s = 0; s < Ssz; ++s)
        m = fmaxf(m, z[((size_t)b * Ssz + s) * Hdim + t]);
    float v = zsrc[(size_t)b * Hdim + t] + b2[t] + m;  // no relu at layer 2
    h2b[(size_t)b * Hdim + t] = __float2bfloat16(v);
}

// ---------------------------------------------------------------------------
extern "C" void kernel_launch(void* const* d_in, const int* in_sizes, int n_in,
                              void* d_out, int out_size, void* d_ws, size_t ws_size,
                              hipStream_t stream) {
    const float* x     = (const float*)d_in[0];  // [N, IN]
    const int*   batch = (const int*)d_in[1];    // [B]
    const int*   nbr1  = (const int*)d_in[2];    // [B, S]
    const int*   nbr2  = (const int*)d_in[3];    // [B, S, S]
    const float* W1    = (const float*)d_in[4];  // [2*IN, H] = [256, 256]
    const float* b1    = (const float*)d_in[5];  // [H]
    const float* W2    = (const float*)d_in[6];  // [2*H, H] = [512, 256]
    const float* b2    = (const float*)d_in[7];  // [H]
    const float* Wout  = (const float*)d_in[8];  // [H, OUT] = [256, 128]
    const float* bout  = (const float*)d_in[9];  // [OUT]
    float* out = (float*)d_out;                  // [B, OUT] f32

    // ---- workspace layout (bytes, all 16B-aligned) ----
    char* ws = (char*)d_ws;
    __hip_bfloat16* xb    = (__hip_bfloat16*)(ws + 0);             // 25,600,000
    __hip_bfloat16* y     = (__hip_bfloat16*)(ws + 25600000);      // 102,400,000
    __hip_bfloat16* Bt1   = (__hip_bfloat16*)(ws + 128000000);     // 131,072  [512,128]
    __hip_bfloat16* Bt2t  = (__hip_bfloat16*)(ws + 128131072);     // 131,072  [256,256]
    __hip_bfloat16* Bt2b  = (__hip_bfloat16*)(ws + 128262144);     // 131,072  [256,256]
    __hip_bfloat16* BtOut = (__hip_bfloat16*)(ws + 128393216);     //  65,536  [128,256]
    __hip_bfloat16* nhb   = (__hip_bfloat16*)(ws + 128458752);     // 16,777,216 [B*S,H]
    __hip_bfloat16* shb   = (__hip_bfloat16*)(ws + 145235968);     //   524,288 [B,H]
    float*          z     = (float*)(ws + 145760256);              // 33,554,432 [B*S,H]
    float*          zsrc  = (float*)(ws + 179314688);              //  1,048,576 [B,H]
    __hip_bfloat16* h2b   = (__hip_bfloat16*)(ws + 180363264);     //    524,288 [B,H]
    // total ~181 MB

    // 1) convert x -> bf16
    {
        int n = Nnod * INF_;
        cvt_bf16_kernel<<<(n + 255) / 256, 256, 0, stream>>>(x, xb, n);
    }
    // 2) pre-transpose weights to bf16 [Nc, K] for the WMMA B operand
    {
        int n = 128 * 256;  // each half of W1
        transpose_bf16_kernel<<<(n + 255) / 256, 256, 0, stream>>>(W1, Bt1, 128, 256);
        transpose_bf16_kernel<<<(n + 255) / 256, 256, 0, stream>>>(
            W1 + 128 * 256, Bt1 + 256 * 128, 128, 256);
        n = 256 * 256;
        transpose_bf16_kernel<<<(n + 255) / 256, 256, 0, stream>>>(W2, Bt2t, 256, 256);
        transpose_bf16_kernel<<<(n + 255) / 256, 256, 0, stream>>>(
            W2 + 256 * 256, Bt2b, 256, 256);
        n = 256 * 128;
        transpose_bf16_kernel<<<(n + 255) / 256, 256, 0, stream>>>(Wout, BtOut, 256, 128);
    }
    // 3) y = xb @ [W1_top | W1_bot]  -> [N, 512] bf16 (L2-resident)
    {
        dim3 grid(Nnod / 32, 512 / 32);
        gemm_bf16_wmma<<<grid, 32, 0, stream>>>(xb, Bt1, nullptr, y, nullptr,
                                                INF_, 512);
    }
    // 4) layer-1 pooled embeddings (neighbors of batch, and batch itself)
    sage1_pool_kernel<<<Bsz * Ssz, 128, 0, stream>>>(y, nbr1, nbr2, b1, nhb);
    sage1_pool_kernel<<<Bsz, 128, 0, stream>>>(y, batch, nbr1, b1, shb);
    // 5) layer-2 projections: z = neigh_h @ W2_bot, zsrc = src_h @ W2_top
    {
        dim3 gz(Bsz * Ssz / 32, Hdim / 32);
        gemm_bf16_wmma<<<gz, 32, 0, stream>>>(nhb, Bt2b, z, nullptr, nullptr,
                                              Hdim, Hdim);
        dim3 gs(Bsz / 32, Hdim / 32);
        gemm_bf16_wmma<<<gs, 32, 0, stream>>>(shb, Bt2t, zsrc, nullptr, nullptr,
                                              Hdim, Hdim);
    }
    // 6) layer-2 combine (max-pool + bias, no relu)
    sage2_combine_kernel<<<Bsz, Hdim, 0, stream>>>(z, zsrc, b2, h2b);
    // 7) output projection: out = h2 @ Wout + bout  -> f32 [B, 128]
    {
        dim3 go(Bsz / 32, OUTD / 32);
        gemm_bf16_wmma<<<go, 32, 0, stream>>>(h2b, BtOut, out, nullptr, bout,
                                              Hdim, OUTD);
    }
}
